// MHSA3D_20667382629155
// MI455X (gfx1250) — matrix operand order repaired and checked by reading, not compiled, not run on hardware
//
#include <hip/hip_runtime.h>

typedef __bf16 bf16_t;
typedef __bf16 v16bf __attribute__((ext_vector_type(16)));
typedef __bf16 v8bf  __attribute__((ext_vector_type(8)));
typedef float  v8f   __attribute__((ext_vector_type(8)));
typedef float  v4f   __attribute__((ext_vector_type(4)));
typedef unsigned int u32x4 __attribute__((ext_vector_type(4)));
typedef int    i32x8 __attribute__((ext_vector_type(8)));
typedef int    i32x4 __attribute__((ext_vector_type(4)));

#define B_      4
#define C_      256
#define F_      8
#define N_      1024
#define DH      64
#define M_ROWS  2048              // C_*F_ (flat row count)
#define SLICES  128               // B_*heads*F_

#ifndef __has_builtin
#define __has_builtin(x) 0
#endif
#if __has_builtin(__builtin_amdgcn_tensor_load_to_lds) && \
    __has_builtin(__builtin_amdgcn_s_wait_tensorcnt)
#define USE_TDM 1
#else
#define USE_TDM 0
#endif

static __device__ __forceinline__ v8f wmma_bf16(v16bf a, v16bf b, v8f c) {
    return __builtin_amdgcn_wmma_f32_16x16x32_bf16(false, a, false, b, (short)0, c,
                                                   false, false);
}

#if USE_TDM
// 2D TDM tile load: tile_x contiguous bf16 per row, tile_y rows, row stride
// stride_e elements.  D# per CDNA5 ISA (6-arg builtin).
static __device__ __forceinline__ void tdm_load_2d(unsigned lds_addr, const void* gptr,
                                                   unsigned tile_x, unsigned tile_y,
                                                   unsigned stride_e) {
    unsigned long long ga = (unsigned long long)(size_t)gptr;
    u32x4 g0;
    g0[0] = 1u;                                            // count=1 (valid user D#)
    g0[1] = lds_addr;                                      // LDS byte address
    g0[2] = (unsigned)(ga & 0xffffffffu);                  // global_addr[31:0]
    g0[3] = (unsigned)((ga >> 32) & 0x01ffffffu) | (2u << 30);  // [56:32] | type=2
    i32x8 g1;
    g1[0] = (int)(1u << 16);                               // data_size=1 (2 bytes)
    g1[1] = (int)(tile_x << 16);                           // tensor_dim0[15:0]
    g1[2] = (int)(tile_y << 16);                           // tensor_dim1[15:0]
    g1[3] = (int)(tile_x << 16);                           // tile_dim0
    g1[4] = (int)tile_y;                                   // tile_dim1 (tile_dim2=0)
    g1[5] = (int)stride_e;                                 // tensor_dim0_stride[31:0]
    g1[6] = 0;
    g1[7] = 0;
    i32x4 z4 = {0, 0, 0, 0};
    i32x8 z8 = {0, 0, 0, 0, 0, 0, 0, 0};
    __builtin_amdgcn_tensor_load_to_lds(g0, g1, z4, z4, z8, 0);
}
#endif

// ---------------------------------------------------------------------------
// pos part of A' buffer:  aT[b][s2][i][64+d] = rel_h + rel_w + rel_t  (bf16)
// ---------------------------------------------------------------------------
__global__ void pos_kernel(const float* __restrict__ rel_h,
                           const float* __restrict__ rel_w,
                           const float* __restrict__ rel_t,
                           bf16_t* __restrict__ aT) {
    int idx = blockIdx.x * 256 + threadIdx.x;      // 4*8*1024*64 = 2097152 exact
    int d  = idx & 63;
    int n  = (idx >> 6) & 1023;
    int fh = idx >> 16;                            // h*8 + f
    int f  = fh & 7;
    int h  = fh >> 3;
    int hh = n >> 5, ww = n & 31;
    float v = rel_h[(h * 64 + d) * 32 + hh]
            + rel_w[(h * 64 + d) * 32 + ww]
            + rel_t[(h * 64 + d) * 8  + f];
    bf16_t bv = (bf16_t)v;
    size_t base = ((size_t)fh * N_ + n) * 128 + 64 + d;
    #pragma unroll
    for (int b = 0; b < 4; ++b)
        aT[base + (size_t)b * (32ull * N_ * 128)] = bv;   // replicate over batch
}

// ---------------------------------------------------------------------------
// Projection GEMM  W(256x256) @ X(256x1024) per (b,f) slice, 3 weights.
//   q -> aT cols [0,64) and kT cols [64,128);  k -> kT cols [0,64);  v -> natural
// ---------------------------------------------------------------------------
__global__ void __launch_bounds__(256) proj_kernel(
        const float* __restrict__ x,
        const float* __restrict__ wq,
        const float* __restrict__ wk,
        const float* __restrict__ wv,
        bf16_t* __restrict__ aT,
        bf16_t* __restrict__ kT,
        bf16_t* __restrict__ vbf) {
    int lane = threadIdx.x & 31;
    int wave = threadIdx.x >> 5;
    int lh = lane >> 4;
    int ll = lane & 15;

    int n0 = blockIdx.x * 16;
    int o0 = (blockIdx.y * 8 + wave) * 16;
    int z  = blockIdx.z;
    int mode  = z % 3;             // 0:q 1:k 2:v
    int slice = z / 3;             // 0..31
    int b = slice >> 3, f = slice & 7;

    const float* w = (mode == 0) ? wq : (mode == 1) ? wk : wv;

    v8f acc = {0, 0, 0, 0, 0, 0, 0, 0};
    for (int c0 = 0; c0 < C_; c0 += 32) {
        const float* wp = w + (size_t)(o0 + ll) * C_ + c0 + (lh << 4);
        const float* xp = x + (((size_t)b * C_ + c0 + (lh << 4)) * F_ + f) * N_ + (n0 + ll);
        v16bf av, bv;
        #pragma unroll
        for (int e = 0; e < 8; ++e) {
            av[e]     = (bf16_t)wp[e];
            av[e + 8] = (bf16_t)wp[8 + e];
        }
        #pragma unroll
        for (int e = 0; e < 16; ++e)
            bv[e] = (bf16_t)xp[(size_t)e * (F_ * N_)];
        acc = wmma_bf16(av, bv, acc);
    }

    int m_base = (o0 + (lh << 3)) * F_ + f;        // d = f + 8r within one hf block
    int hf = m_base >> 6;
    size_t s2 = (size_t)b * 32 + hf;
    if (mode == 0) {
        bf16_t* ap = aT + (s2 * N_ + n0 + ll) * 128 + f;
        bf16_t* kp = kT + (s2 * N_ + n0 + ll) * 128 + 64 + f;
        #pragma unroll
        for (int r = 0; r < 8; ++r) {
            bf16_t vv = (bf16_t)acc[r];
            ap[8 * r] = vv;
            kp[8 * r] = vv;
        }
    } else if (mode == 1) {
        bf16_t* kp = kT + (s2 * N_ + n0 + ll) * 128 + f;
        #pragma unroll
        for (int r = 0; r < 8; ++r) kp[8 * r] = (bf16_t)acc[r];
    } else {
        bf16_t* vp = vbf + ((size_t)b * M_ROWS + m_base) * N_ + n0 + ll;
        #pragma unroll
        for (int r = 0; r < 8; ++r) vp[(size_t)(8 * r) * N_] = (bf16_t)acc[r];
    }
}

// ---------------------------------------------------------------------------
// Fused flash attention.  32 queries per wave (two 16-row tiles sharing each
// staged K'/V tile), double-buffered TDM staging pipelined against WMMA.
// grid = (4 query-blocks, 128 slices), block = 256 (8 waves)
// ---------------------------------------------------------------------------
__global__ void __launch_bounds__(256) attn_kernel(
        const bf16_t* __restrict__ aT,
        const bf16_t* __restrict__ kT,
        const bf16_t* __restrict__ vbf,
        float* __restrict__ out) {
    // LDS: K' bufs 2x8KB @0, V bufs 2x4KB @16KB, P 16KB @24KB  (40KB total)
    __shared__ __align__(32) unsigned int smem_u[10240];
    bf16_t* smemKb = (bf16_t*)smem_u;              // + buf*4096 elems
    bf16_t* smemVb = (bf16_t*)smem_u + 8192;       // + buf*2048 elems
    bf16_t* smemP  = (bf16_t*)smem_u + 12288;      // + (wave*2+iA)*512 elems
    float*  smemO  = (float*)smem_u;               // output staging (reuse)

    int tid  = threadIdx.x;
    int lane = tid & 31, wave = tid >> 5;
    int lh = lane >> 4, ll = lane & 15;

    int s  = blockIdx.y;
    int b  = s >> 5;
    int m0 = (s & 31) << 6;
    const bf16_t* kTs = kT + (size_t)s * N_ * 128;
    const bf16_t* vbs = vbf + ((size_t)b * M_ROWS + m0) * N_;
    int i0 = blockIdx.x * 256 + wave * 32;

    // ---- A' operands for both 16-row tiles (resident) ----
    v16bf aQ[2][4];
    #pragma unroll
    for (int iA = 0; iA < 2; ++iA) {
        const bf16_t* ap = aT + ((size_t)s * N_ + i0 + iA * 16 + ll) * 128 + (lh << 4);
        #pragma unroll
        for (int t = 0; t < 4; ++t) {
            v8bf lo = *(const v8bf*)(ap + t * 32);
            v8bf hi = *(const v8bf*)(ap + t * 32 + 8);
            aQ[iA][t] = __builtin_shufflevector(lo, hi, 0, 1, 2, 3, 4, 5, 6, 7,
                                                8, 9, 10, 11, 12, 13, 14, 15);
        }
    }

    float mrow[2][8], lrow[2][8];
    v8f acc[2][4];
    #pragma unroll
    for (int iA = 0; iA < 2; ++iA) {
        #pragma unroll
        for (int r = 0; r < 8; ++r) { mrow[iA][r] = -1e30f; lrow[iA][r] = 0.0f; }
        #pragma unroll
        for (int t = 0; t < 4; ++t) acc[iA][t] = (v8f){0, 0, 0, 0, 0, 0, 0, 0};
    }

    const float sc = 0.125f * 1.44269504088896340736f;    // dh^-0.5 * log2(e)

#if USE_TDM
    unsigned kaddr = (unsigned)(size_t)(void*)smemKb;     // LDS byte addresses
    unsigned vaddr = (unsigned)(size_t)(void*)smemVb;
    if (wave == 0) {                                      // prologue: tile 0 -> buf 0
        tdm_load_2d(kaddr, kTs, 128, 32, 128);
        tdm_load_2d(vaddr, vbs, 32, 64, N_);
    }
#endif

    for (int it = 0; it < 32; ++it) {
        int buf = it & 1;
        bf16_t* kS = smemKb + buf * 4096;
        bf16_t* vS = smemVb + buf * 2048;
#if USE_TDM
        if (wave == 0) {
            // prefetch next tile into the other buffer (wraps harmlessly at end)
            int jn = ((it + 1) & 31) * 32;
            tdm_load_2d(kaddr + (buf ^ 1) * 8192u, kTs + (size_t)jn * 128, 128, 32, 128);
            tdm_load_2d(vaddr + (buf ^ 1) * 4096u, vbs + jn, 32, 64, N_);
            __builtin_amdgcn_s_wait_tensorcnt((short)2);  // current tile complete
        }
        __syncthreads();                                   // tile visible to all waves
#else
        int j0 = it * 32;
        u32x4* sk4 = (u32x4*)(smem_u + buf * 2048);        // K buf (u32 units: 2048)
        u32x4* sv4 = (u32x4*)(smem_u + 4096 + buf * 1024); // V buf
        for (int idx = tid; idx < 768; idx += 256) {
            if (idx < 512) {
                int j = idx >> 4, c = idx & 15;
                sk4[idx] = ((const u32x4*)(kTs + (size_t)(j0 + j) * 128))[c];
            } else {
                int i2 = idx - 512;
                int d = i2 >> 2, c = i2 & 3;
                sv4[i2] = ((const u32x4*)(vbs + (size_t)d * N_ + j0))[c];
            }
        }
        __syncthreads();
#endif

        #pragma unroll
        for (int iA = 0; iA < 2; ++iA) {
            // ---- energy: two 16x16 key sub-tiles, K=128 ----
            v8f c0v = {0, 0, 0, 0, 0, 0, 0, 0};
            v8f c1v = {0, 0, 0, 0, 0, 0, 0, 0};
            #pragma unroll
            for (int t = 0; t < 4; ++t) {
                v16bf b0 = *(const v16bf*)(kS + (size_t)ll * 128 + t * 32 + (lh << 4));
                v16bf b1 = *(const v16bf*)(kS + (size_t)(16 + ll) * 128 + t * 32 + (lh << 4));
                c0v = wmma_bf16(aQ[iA][t], b0, c0v);
                c1v = wmma_bf16(aQ[iA][t], b1, c1v);
            }

            // ---- online softmax ----
            bf16_t* myP = smemP + (wave * 2 + iA) * 512;   // wave-private region
            #pragma unroll
            for (int r = 0; r < 8; ++r) {
                float x0 = c0v[r] * sc, x1 = c1v[r] * sc;
                float tmax = fmaxf(x0, x1);
                #pragma unroll
                for (int off = 1; off < 16; off <<= 1)
                    tmax = fmaxf(tmax, __shfl_xor(tmax, off, 32));
                float mnew = fmaxf(mrow[iA][r], tmax);
                float p0 = exp2f(x0 - mnew);
                float p1 = exp2f(x1 - mnew);
                float rs = p0 + p1;
                #pragma unroll
                for (int off = 1; off < 16; off <<= 1)
                    rs += __shfl_xor(rs, off, 32);
                float corr = exp2f(mrow[iA][r] - mnew);
                lrow[iA][r] = lrow[iA][r] * corr + rs;
                mrow[iA][r] = mnew;
                #pragma unroll
                for (int t = 0; t < 4; ++t) acc[iA][t][r] *= corr;
                int prow = (lh << 3) + r;
                myP[prow * 32 + ll]      = (bf16_t)p0;
                myP[prow * 32 + 16 + ll] = (bf16_t)p1;
            }
            // same-wave LDS ops are in-order: no block barrier needed here.

            // ---- PV: P(16x32) as A, V sub-tiles as B ----
            v8bf plo = *(const v8bf*)(myP + ll * 32 + (lh << 4));
            v8bf phi = *(const v8bf*)(myP + ll * 32 + 8 + (lh << 4));
            v16bf pa = __builtin_shufflevector(plo, phi, 0, 1, 2, 3, 4, 5, 6, 7,
                                               8, 9, 10, 11, 12, 13, 14, 15);
            #pragma unroll
            for (int t = 0; t < 4; ++t) {
                v16bf bv = *(const v16bf*)(vS + (t * 16 + ll) * 32 + (lh << 4));
                acc[iA][t] = wmma_bf16(pa, bv, acc[iA][t]);
            }
        }
        __syncthreads();                           // all waves done with this tile
    }

#if USE_TDM
    if (wave == 0) __builtin_amdgcn_s_wait_tensorcnt((short)0);  // drain prefetch
#endif
    __syncthreads();                               // safe to reuse staging LDS

    // ---- finalize: 1/l, per-wave LDS transpose, coalesced 64B stores ----
    float* myO = smemO + wave * (64 * 16);         // wave-private 4KB
    #pragma unroll
    for (int iA = 0; iA < 2; ++iA) {
        #pragma unroll
        for (int r = 0; r < 8; ++r) {
            float inv = 1.0f / lrow[iA][r];
            #pragma unroll
            for (int t = 0; t < 4; ++t) acc[iA][t][r] *= inv;
        }
        #pragma unroll
        for (int t = 0; t < 4; ++t)
            #pragma unroll
            for (int r = 0; r < 8; ++r)
                myO[(t * 16 + ll) * 16 + (lh << 3) + r] = acc[iA][t][r];
        // same-wave in-order LDS: transpose visible to this wave's lanes
        float* ob = out + ((size_t)b * M_ROWS + m0) * N_ + i0 + iA * 16;
        #pragma unroll
        for (int rr = 0; rr < 2; ++rr) {
            int d = lane * 2 + rr;
            const v4f* src = (const v4f*)(myO + d * 16);
            v4f* dstp = (v4f*)(ob + (size_t)d * N_);
            #pragma unroll
            for (int qd = 0; qd < 4; ++qd) dstp[qd] = src[qd];
        }
    }
}

// ---------------------------------------------------------------------------
extern "C" void kernel_launch(void* const* d_in, const int* in_sizes, int n_in,
                              void* d_out, int out_size, void* d_ws, size_t ws_size,
                              hipStream_t stream) {
    const float* x    = (const float*)d_in[0];
    const float* wq   = (const float*)d_in[1];
    const float* wk   = (const float*)d_in[2];
    const float* wv   = (const float*)d_in[3];
    const float* relh = (const float*)d_in[4];
    const float* relw = (const float*)d_in[5];
    const float* relt = (const float*)d_in[6];
    float* out = (float*)d_out;

    char* ws = (char*)d_ws;
    const size_t AT_BYTES = (size_t)SLICES * N_ * 128 * sizeof(bf16_t);  // 32 MB
    bf16_t* aTb = (bf16_t*)(ws);                                         // [s][i][128]
    bf16_t* kTb = (bf16_t*)(ws + AT_BYTES);                              // [s][j][128]
    bf16_t* vbf = (bf16_t*)(ws + 2 * AT_BYTES);                          // [B][2048][N]

    pos_kernel <<<8192, 256, 0, stream>>>(relh, relw, relt, aTb);
    proj_kernel<<<dim3(64, 2, 96), 256, 0, stream>>>(x, wq, wk, wv, aTb, kTb, vbf);
    attn_kernel<<<dim3(4, 128), 256, 0, stream>>>(aTb, kTb, vbf, out);
}
